// AttentionBlock_64682207477810
// MI455X (gfx1250) — matrix operand order repaired and checked by reading, not compiled, not run on hardware
//
#include <hip/hip_runtime.h>
#include <hip/hip_bf16.h>

// ---------------------------------------------------------------------------
// AttentionBlock for MI455X (gfx1250, wave32, WMMA 16x16x32 f16).
// b=8, c=128, p=2048, h=4 (c*h=512).
//
// All WMMA fragments are contiguous 128-bit loads:
//   A-frag (16x32): lane m reads halves {ka..ka+15}, ka=0/16
//   B-frag (32x16): lane n reads halves {kb..kb+15}, kb=0/16
// Staging layouts make the per-lane K run contiguous:
//   xnT[b][p][c]      : proj B operand
//   Whq/Whk/Whv/Who   : f16 weights (pre-cast) -> A operands
//   Qt[bh][p][c]      : QK^T A operand
//   Kt[bh][p][c]      : QK^T B operand (row j, contiguous c)
//   Vc[bh][c][p]      : PV   B operand (row c, contiguous j)
//   AtB[b][p][c*h]    : outproj B operand (contiguous composite K)
// Attention kernel stages the shared K/V j-tiles in LDS once per block
// (8 waves consume identical B operands), cutting global B traffic 8x.
// ---------------------------------------------------------------------------

typedef __attribute__((ext_vector_type(16))) _Float16 v16h;
typedef __attribute__((ext_vector_type(8)))  _Float16 v8h;
typedef __attribute__((ext_vector_type(8)))  float    v8f;

constexpr int B_  = 8;
constexpr int C_  = 128;
constexpr int P_  = 2048;
constexpr int H_  = 4;
constexpr int CH_ = C_ * H_;   // 512

// A-fragment: two 16B chunks at p[0..7] and p[8..15] (caller adds lane-half
// offset (lane&16)?16:0 into p). Works for global and LDS pointers.
__device__ __forceinline__ v16h ldA(const _Float16* p) {
    const v8h* q = (const v8h*)p;
    v8h lo = q[0];
    v8h hi = q[1];  // +8 halves
    return __builtin_shufflevector(lo, hi, 0, 1, 2, 3, 4, 5, 6, 7,
                                           8, 9, 10, 11, 12, 13, 14, 15);
}
// B-fragment: 16 contiguous halves (32B aligned at all call sites).
__device__ __forceinline__ v16h ldB(const _Float16* p) {
    return *(const v16h*)p;
}
// C/D row for accumulator element r.
__device__ __forceinline__ int mD(int lane, int r) {
    return r + ((lane & 16) ? 8 : 0);
}

// --------------------------- 0) weight f32 -> f16 ---------------------------
__global__ void __launch_bounds__(256)
f16cast_kernel(const float* __restrict__ src, _Float16* __restrict__ dst) {
    int idx = blockIdx.x * 256 + threadIdx.x;
    dst[idx] = (_Float16)src[idx];
}

// --------------------------- 1) batchnorm stats -----------------------------
__global__ void __launch_bounds__(256)
bn_stats_kernel(const float* __restrict__ inp, float* __restrict__ mean,
                float* __restrict__ rstd) {
    __shared__ float s1[256];
    __shared__ float s2[256];
    const int ch  = blockIdx.x;
    const int tid = threadIdx.x;
    float a = 0.f, b = 0.f;
    for (int idx = tid; idx < B_ * P_; idx += 256) {
        int bi = idx / P_, pi = idx % P_;
        float v = inp[((size_t)bi * C_ + ch) * P_ + pi];
        a += v;
        b += v * v;
    }
    s1[tid] = a; s2[tid] = b;
    __syncthreads();
    for (int off = 128; off > 0; off >>= 1) {
        if (tid < off) { s1[tid] += s1[tid + off]; s2[tid] += s2[tid + off]; }
        __syncthreads();
    }
    if (tid == 0) {
        const float inv_n = 1.0f / (float)(B_ * P_);
        float m   = s1[0] * inv_n;
        float var = s2[0] * inv_n - m * m;
        mean[ch] = m;
        rstd[ch] = rsqrtf(var + 1e-5f);
    }
}

// ---------------- 2) normalize -> f16, transposed to [b][p][c] --------------
__global__ void __launch_bounds__(256)
bn_norm_kernel(const float* __restrict__ inp, const float* __restrict__ mean,
               const float* __restrict__ rstd, const float* __restrict__ gamma,
               const float* __restrict__ beta, _Float16* __restrict__ xnT) {
    size_t idx = (size_t)blockIdx.x * 256 + threadIdx.x;  // over [b][c][p]
    int pi = (int)(idx % P_);
    int ci = (int)((idx / P_) % C_);
    int bi = (int)(idx / ((size_t)C_ * P_));
    float v = (inp[idx] - mean[ci]) * rstd[ci] * gamma[ci] + beta[ci];
    xnT[((size_t)bi * P_ + pi) * C_ + ci] = (_Float16)v;
}

// --------------------------- 3) q/k/v projections ---------------------------
// grid.x = B*(CH/16)*(P/16)/8, grid.y = {q,k,v}; 8 waves/block, tile/wave.
__global__ void __launch_bounds__(256)
proj_kernel(const _Float16* __restrict__ xnT,
            const _Float16* __restrict__ Whq, const float* __restrict__ bq,
            const _Float16* __restrict__ Whk, const float* __restrict__ bk,
            const _Float16* __restrict__ Whv, const float* __restrict__ bv,
            _Float16* __restrict__ Qt, _Float16* __restrict__ Kt,
            _Float16* __restrict__ Vc) {
    const int lane = threadIdx.x & 31;
    const int wave = threadIdx.x >> 5;
    const int NT = P_ / 16;    // 128
    const int MT = CH_ / 16;   // 32
    int tile = blockIdx.x * 8 + wave;
    int bi   = tile / (MT * NT);
    int rem  = tile % (MT * NT);
    int m0   = (rem / NT) * 16;
    int n0   = (rem % NT) * 16;

    const int which = blockIdx.y;
    const _Float16* W    = (which == 0) ? Whq : (which == 1) ? Whk : Whv;
    const float*    bias = (which == 0) ? bq  : (which == 1) ? bk  : bv;

    const int ma = lane & 15;
    const int nb = lane & 15;
    const int ha = (lane & 16) ? 8 : 0;
    const int kb = (lane & 16) ? 16 : 0;

    const _Float16* Wrow = W + (size_t)(m0 + ma) * C_ + kb;
    const _Float16* Xrow = xnT + ((size_t)bi * P_ + n0 + nb) * C_ + kb;

    v8f acc = {0.f, 0.f, 0.f, 0.f, 0.f, 0.f, 0.f, 0.f};
#pragma unroll
    for (int kk = 0; kk < C_ / 32; ++kk) {
        v16h a = ldA(Wrow + kk * 32);
        v16h b = ldB(Xrow + kk * 32);
        acc = __builtin_amdgcn_wmma_f32_16x16x32_f16(false, a, false, b,
                                                     (short)0, acc, false, false);
    }

    // epilogue: bias + scatter (uniform `which` branch lifted out of the loop)
    if (which == 0) {
#pragma unroll
        for (int r = 0; r < 8; ++r) {
            int o = m0 + mD(lane, r);
            int ci = o / H_, hi = o % H_;
            size_t bh = (size_t)bi * H_ + hi;
            Qt[(bh * P_ + n0 + nb) * C_ + ci] = (_Float16)(acc[r] + bias[o]);
        }
    } else if (which == 1) {
#pragma unroll
        for (int r = 0; r < 8; ++r) {
            int o = m0 + mD(lane, r);
            int ci = o / H_, hi = o % H_;
            size_t bh = (size_t)bi * H_ + hi;
            Kt[(bh * P_ + n0 + nb) * C_ + ci] = (_Float16)(acc[r] + bias[o]);
        }
    } else {
#pragma unroll
        for (int r = 0; r < 8; ++r) {
            int o = m0 + mD(lane, r);
            int ci = o / H_, hi = o % H_;
            size_t bh = (size_t)bi * H_ + hi;
            Vc[(bh * C_ + ci) * P_ + n0 + nb] = (_Float16)(acc[r] + bias[o]);
        }
    }
}

// --------------------------- 4) flash attention -----------------------------
// grid.x = B*H*(P/128); block = 256 (8 waves); wave owns 16 query points.
// K/V j-tiles are staged cooperatively in LDS once per block per iteration.
__global__ void __launch_bounds__(256)
attn_kernel(const _Float16* __restrict__ Qt, const _Float16* __restrict__ Kt,
            const _Float16* __restrict__ Vc, _Float16* __restrict__ AtB) {
    __shared__ __align__(16) _Float16 Ksh[32 * 128];      // [j_loc][c]  8 KB
    __shared__ __align__(16) _Float16 Vsh[128 * 32];      // [c][j_loc]  8 KB
    __shared__ __align__(16) _Float16 ldsP[8][16 * 40];   // per-wave P tile

    const int tid   = threadIdx.x;
    const int lane  = tid & 31;
    const int wave  = tid >> 5;
    const int bh    = blockIdx.x / (P_ / 128);
    const int itile = blockIdx.x % (P_ / 128);
    const int i0    = itile * 128 + wave * 16;
    const int bi    = bh / H_;
    const int hi    = bh % H_;

    const _Float16* Qb = Qt + (size_t)bh * P_ * C_;
    const _Float16* Kb = Kt + (size_t)bh * P_ * C_;
    const _Float16* Vb = Vc + (size_t)bh * C_ * P_;

    const int ma = lane & 15;
    const int nb = lane & 15;
    const int ha = (lane & 16) ? 8 : 0;
    const int kb = (lane & 16) ? 16 : 0;

    // cooperative-fill mapping: 256 threads x one v16h (32B) chunk per tile
    const int krow = tid >> 3, kcol = (tid & 7) * 16;   // K: 32 rows x 128 halves
    const int vrow = tid >> 1, vcol = (tid & 1) * 16;   // V: 128 rows x 32 halves

    // Q A-tiles: 4 chunks of K=32 over c=128, two b128 loads each
    const _Float16* Qrow = Qb + (size_t)(i0 + ma) * C_ + kb;
    v16h Aq[4];
#pragma unroll
    for (int kk = 0; kk < 4; ++kk) Aq[kk] = ldA(Qrow + kk * 32);

    float rmax[8], rsum[8];
    v8f   O[8];
#pragma unroll
    for (int r = 0; r < 8; ++r) { rmax[r] = -1e30f; rsum[r] = 0.f; }
#pragma unroll
    for (int t = 0; t < 8; ++t) O[t] = (v8f){0.f,0.f,0.f,0.f,0.f,0.f,0.f,0.f};

    const float scale = 0.08838834764831845f;  // c^-0.5, c=128
    _Float16* pslab = &ldsP[wave][0];

    for (int j0 = 0; j0 < P_; j0 += 32) {
        // ---- stage K/V tiles for this j chunk (shared by all 8 waves) ----
        __syncthreads();  // previous iteration's LDS consumers are done
        *(v16h*)&Ksh[krow * 128 + kcol] =
            *(const v16h*)(Kb + (size_t)(j0 + krow) * C_ + kcol);
        *(v16h*)&Vsh[vrow * 32 + vcol] =
            *(const v16h*)(Vb + (size_t)vrow * P_ + j0 + vcol);
        if (j0 + 32 < P_)
            __builtin_prefetch(Kb + (size_t)(j0 + 32 + krow) * C_ + kcol, 0, 0);
        __syncthreads();

        // ---- S = Q K^T for columns [j0, j0+32): two 16x16 tiles ----
        const _Float16* Kr0 = &Ksh[(nb) * 128 + kb];
        const _Float16* Kr1 = &Ksh[(16 + nb) * 128 + kb];
        v8f S0 = (v8f){0.f,0.f,0.f,0.f,0.f,0.f,0.f,0.f};
        v8f S1 = (v8f){0.f,0.f,0.f,0.f,0.f,0.f,0.f,0.f};
#pragma unroll
        for (int kk = 0; kk < 4; ++kk) {
            v16h b0 = ldB(Kr0 + kk * 32);
            v16h b1 = ldB(Kr1 + kk * 32);
            S0 = __builtin_amdgcn_wmma_f32_16x16x32_f16(false, Aq[kk], false, b0,
                                                        (short)0, S0, false, false);
            S1 = __builtin_amdgcn_wmma_f32_16x16x32_f16(false, Aq[kk], false, b1,
                                                        (short)0, S1, false, false);
        }

        // ---- online softmax (row reductions stay inside 16-lane halves) ----
        float f[8];
#pragma unroll
        for (int r = 0; r < 8; ++r) {
            float s0 = S0[r] * scale, s1 = S1[r] * scale;
            float mx = fmaxf(s0, s1);
            mx = fmaxf(mx, __shfl_xor(mx, 1));
            mx = fmaxf(mx, __shfl_xor(mx, 2));
            mx = fmaxf(mx, __shfl_xor(mx, 4));
            mx = fmaxf(mx, __shfl_xor(mx, 8));
            float nm = fmaxf(rmax[r], mx);
            f[r]    = __expf(rmax[r] - nm);
            rmax[r] = nm;
            float p0 = __expf(s0 - nm), p1 = __expf(s1 - nm);
            S0[r] = p0; S1[r] = p1;
            float ls = p0 + p1;
            ls += __shfl_xor(ls, 1);
            ls += __shfl_xor(ls, 2);
            ls += __shfl_xor(ls, 4);
            ls += __shfl_xor(ls, 8);
            rsum[r] = rsum[r] * f[r] + ls;
        }
#pragma unroll
        for (int t = 0; t < 8; ++t)
#pragma unroll
            for (int r = 0; r < 8; ++r) O[t][r] *= f[r];

        // ---- transpose P (D layout -> A layout) through LDS ----
#pragma unroll
        for (int r = 0; r < 8; ++r) {
            int m = mD(lane, r);
            pslab[m * 40 + nb]      = (_Float16)S0[r];
            pslab[m * 40 + 16 + nb] = (_Float16)S1[r];
        }
        __syncthreads();
        v16h pa = ldA(pslab + ma * 40 + kb);   // two ds_load_b128

        // ---- O += P * V : 8 tiles cover the 128 output channels ----
#pragma unroll
        for (int t = 0; t < 8; ++t) {
            v16h bv = ldB(&Vsh[(t * 16 + nb) * 32 + kb]);
            O[t] = __builtin_amdgcn_wmma_f32_16x16x32_f16(false, pa, false, bv,
                                                          (short)0, O[t], false, false);
        }
    }

    // ---- normalize, store att to AtB[b][p][ci*h+hi] (outproj B layout) ----
#pragma unroll
    for (int r = 0; r < 8; ++r) {
        float inv = 1.0f / rsum[r];
        int pi = i0 + mD(lane, r);
        size_t rowbase = ((size_t)bi * P_ + pi) * CH_ + hi;
#pragma unroll
        for (int t = 0; t < 8; ++t) {
            int ci = t * 16 + nb;
            AtB[rowbase + (size_t)ci * H_] = (_Float16)(O[t][r] * inv);
        }
    }
}

// --------------------------- 5) output projection ---------------------------
__global__ void __launch_bounds__(256)
outproj_kernel(const _Float16* __restrict__ AtB, const _Float16* __restrict__ Who,
               const float* __restrict__ bo, const float* __restrict__ inp,
               float* __restrict__ out) {
    const int lane = threadIdx.x & 31;
    const int wave = threadIdx.x >> 5;
    const int NT = P_ / 16;   // 128
    const int MT = C_ / 16;   // 8
    int tile = blockIdx.x * 8 + wave;
    int bi   = tile / (MT * NT);
    int rem  = tile % (MT * NT);
    int m0   = (rem / NT) * 16;
    int n0   = (rem % NT) * 16;

    const int ma = lane & 15;
    const int nb = lane & 15;
    const int ha = (lane & 16) ? 8 : 0;
    const int kb = (lane & 16) ? 16 : 0;

    const _Float16* Wrow = Who + (size_t)(m0 + ma) * CH_ + kb;
    const _Float16* Brow = AtB + ((size_t)bi * P_ + n0 + nb) * CH_ + kb;

    v8f acc = {0.f, 0.f, 0.f, 0.f, 0.f, 0.f, 0.f, 0.f};
#pragma unroll
    for (int kk = 0; kk < CH_ / 32; ++kk) {   // K = 512, 16 chunks
        v16h a = ldA(Wrow + kk * 32);
        v16h b = ldB(Brow + kk * 32);
        acc = __builtin_amdgcn_wmma_f32_16x16x32_f16(false, a, false, b,
                                                     (short)0, acc, false, false);
    }

#pragma unroll
    for (int r = 0; r < 8; ++r) {
        int o = m0 + mD(lane, r);
        size_t idx = (size_t)bi * C_ * P_ + (size_t)o * P_ + n0 + nb;
        out[idx] = acc[r] + bo[o] + inp[idx];   // + residual
    }
}

// ---------------------------------------------------------------------------
extern "C" void kernel_launch(void* const* d_in, const int* in_sizes, int n_in,
                              void* d_out, int out_size, void* d_ws, size_t ws_size,
                              hipStream_t stream) {
    const float* inp   = (const float*)d_in[0];
    const float* gamma = (const float*)d_in[1];
    const float* beta  = (const float*)d_in[2];
    const float* Wq    = (const float*)d_in[3];
    const float* bq    = (const float*)d_in[4];
    const float* Wk    = (const float*)d_in[5];
    const float* bk    = (const float*)d_in[6];
    const float* Wv    = (const float*)d_in[7];
    const float* bv    = (const float*)d_in[8];
    const float* Wo    = (const float*)d_in[9];
    const float* bo    = (const float*)d_in[10];
    float* out = (float*)d_out;

    char* ws = (char*)d_ws;
    const size_t nW   = (size_t)CH_ * C_;                              // 65536
    const size_t szW  = nW * sizeof(_Float16);                         // 128 KB
    const size_t szXN = (size_t)B_ * P_ * C_ * sizeof(_Float16);       // 4 MB
    const size_t szT  = (size_t)B_ * H_ * P_ * C_ * sizeof(_Float16);  // 16 MB

    float*    mean = (float*)(ws + 0);
    float*    rstd = (float*)(ws + 512);
    size_t off = 1024;
    _Float16* Whq = (_Float16*)(ws + off); off += szW;
    _Float16* Whk = (_Float16*)(ws + off); off += szW;
    _Float16* Whv = (_Float16*)(ws + off); off += szW;
    _Float16* Who = (_Float16*)(ws + off); off += szW;
    _Float16* xnT = (_Float16*)(ws + off); off += szXN;
    _Float16* Qt  = (_Float16*)(ws + off); off += szT;
    _Float16* Kt  = (_Float16*)(ws + off); off += szT;
    _Float16* Vc  = (_Float16*)(ws + off); off += szT;
    _Float16* AtB = (_Float16*)(ws + off); off += szT;

    const int wgrid = (int)(nW / 256);
    f16cast_kernel<<<wgrid, 256, 0, stream>>>(Wq, Whq);
    f16cast_kernel<<<wgrid, 256, 0, stream>>>(Wk, Whk);
    f16cast_kernel<<<wgrid, 256, 0, stream>>>(Wv, Whv);
    f16cast_kernel<<<wgrid, 256, 0, stream>>>(Wo, Who);

    bn_stats_kernel<<<C_, 256, 0, stream>>>(inp, mean, rstd);
    bn_norm_kernel<<<(B_ * C_ * P_) / 256, 256, 0, stream>>>(inp, mean, rstd,
                                                             gamma, beta, xnT);
    proj_kernel<<<dim3((B_ * (CH_ / 16) * (P_ / 16)) / 8, 3), 256, 0, stream>>>(
        xnT, Whq, bq, Whk, bk, Whv, bv, Qt, Kt, Vc);
    attn_kernel<<<B_ * H_ * (P_ / 128), 256, 0, stream>>>(Qt, Kt, Vc, AtB);
    outproj_kernel<<<(B_ * (C_ / 16) * (P_ / 16)) / 8, 256, 0, stream>>>(
        AtB, Who, bo, inp, out);
}
